// EnDeModel_13237089206430
// MI455X (gfx1250) — matrix operand order repaired and checked by reading, not compiled, not run on hardware
//
#include <hip/hip_runtime.h>
#include <hip/hip_bf16.h>

#define Hh    2048
#define R3H   6144     // 3*H
#define TSTEPS 365
#define FSTEPS 90
#define KSPLIT 4
#define GPS   12288    // partial-buffer stride (rows per split buffer)

typedef __attribute__((ext_vector_type(16))) __bf16 v16bf;
typedef __attribute__((ext_vector_type(8)))  float  v8f;

__device__ __forceinline__ unsigned short f2bf(float f) {
    unsigned int u = __float_as_uint(f);
    u += 0x7FFFu + ((u >> 16) & 1u);          // round-to-nearest-even
    return (unsigned short)(u >> 16);
}
__device__ __forceinline__ float sigm(float x) { return 1.0f / (1.0f + expf(-x)); }
__device__ __forceinline__ float gsum(const float* __restrict__ gp, int i) {
    return (gp[i] + gp[GPS + i]) + (gp[2 * GPS + i] + gp[3 * GPS + i]);
}

// ---------------------------------------------------------------------------
// f32 -> bf16 weight conversion
// ---------------------------------------------------------------------------
__global__ __launch_bounds__(256) void f32_to_bf16_k(
    const float* __restrict__ src, unsigned short* __restrict__ dst, int n)
{
    int i = blockIdx.x * 256 + threadIdx.x;
    if (i < n) dst[i] = f2bf(src[i]);
}

// ---------------------------------------------------------------------------
// Encoder input-projection collapse:
//   A[row] = enc_Wih[row,:] @ Wb[:,0]
//   B[row] = enc_Wih[row,:] @ bb + enc_bih[row]
// ---------------------------------------------------------------------------
__global__ __launch_bounds__(256) void enc_input_proj(
    const float* __restrict__ Wih, const float* __restrict__ Wb,
    const float* __restrict__ bb,  const float* __restrict__ bih,
    float* __restrict__ A, float* __restrict__ B)
{
    int wave = threadIdx.x >> 5, lane = threadIdx.x & 31;
    int row = blockIdx.x * 8 + wave;
    if (row >= R3H) return;
    const float* wr = Wih + (size_t)row * Hh;
    float sa = 0.f, sb = 0.f;
    for (int k = lane; k < Hh; k += 32) {
        float w = wr[k];
        sa = fmaf(w, Wb[k], sa);
        sb = fmaf(w, bb[k], sb);
    }
    for (int off = 16; off; off >>= 1) {
        sa += __shfl_xor(sa, off, 32);
        sb += __shfl_xor(sb, off, 32);
    }
    if (lane == 0) { A[row] = sa; B[row] = sb + bih[row]; }
}

__global__ void init_state(float* __restrict__ h,
                           unsigned short* __restrict__ hb,
                           unsigned short* __restrict__ xb)
{
    int i = blockIdx.x * 256 + threadIdx.x;
    if (i < Hh) { h[i] = 0.f; hb[i] = 0; xb[i] = 0; }
}

// ---------------------------------------------------------------------------
// bf16 mat-vec via v_wmma_f32_16x16x32_bf16, K split 4 ways.
// Wave (strip, split): 16-row strip, K range [split*512, split*512+512).
// A tile (16x32 bf16, row-major W):
//   lanes 0-15  : row m, K=0..7  (bytes 0..15)  + K=16..23 (bytes 32..47)
//   lanes 16-31 : row m, K=8..15 (bytes 16..31) + K=24..31 (bytes 48..63)
// B column N=0 only: lanes 0/16 read the vector from LDS[0..4KB);
// all other lanes read a zeroed LDS region [4KB..8KB) -> no cndmask, no
// WMMA->VALU WAR hazard NOPs in the hot loop.
// D column 0: lane 0 -> rows m0..m0+7 (VGPR 0..7), lane 16 -> rows m0+8..+15.
// Strips 0..383 use (W0,v0); strips >=384 use (W1,v1) (decoder stack).
// ---------------------------------------------------------------------------
__global__ __launch_bounds__(256) void matvec_bf16_wmma(
    const unsigned short* __restrict__ W0, const unsigned short* __restrict__ v0,
    const unsigned short* __restrict__ W1, const unsigned short* __restrict__ v1,
    float* __restrict__ gpart)
{
    __shared__ uint4 lds[512];                       // 4 KB vector + 4 KB zeros
    int gwave0 = blockIdx.x * 8;
    int strip0 = gwave0 >> 2;
    bool half2 = (strip0 >= 384);
    const unsigned short* W   = half2 ? W1 : W0;
    const unsigned short* vec = half2 ? v1 : v0;

    lds[threadIdx.x]       = ((const uint4*)vec)[threadIdx.x];
    lds[256 + threadIdx.x] = make_uint4(0u, 0u, 0u, 0u);
    __syncthreads();

    int wave  = threadIdx.x >> 5;
    int lane  = threadIdx.x & 31;
    int gwave = gwave0 + wave;
    int strip = gwave >> 2;                          // 4 splits per strip
    int split = gwave & 3;
    int rowBase  = strip * 16;                       // global output row
    int rowLocal = rowBase - (half2 ? R3H : 0);      // row within W
    int m  = lane & 15;
    bool hi  = lane >= 16;
    bool act = (lane & 15) == 0;
    const unsigned short* Wrow =
        W + (size_t)(rowLocal + m) * Hh + split * 512;   // K offset for split
    // uint4 index into LDS: active lanes -> vector, others -> zero region
    int bbase = (act ? 0 : 256) + split * 64 + (hi ? 1 : 0);

    v8f c = {0.f, 0.f, 0.f, 0.f, 0.f, 0.f, 0.f, 0.f};
    const uint4* ldsq = lds;

    #pragma unroll 4
    for (int j = 0; j < 16; ++j) {                   // 16 K-chunks of 32
        v16bf a, b;
        uint4* ap = (uint4*)&a;
        uint4* bp = (uint4*)&b;
        int aOff = j * 32 + (hi ? 8 : 0);            // ushort index within row
        ap[0] = *(const uint4*)(Wrow + aOff);
        ap[1] = *(const uint4*)(Wrow + aOff + 16);
        bp[0] = ldsq[bbase + j * 4];
        bp[1] = ldsq[bbase + j * 4 + 2];
        c = __builtin_amdgcn_wmma_f32_16x16x32_bf16(
                false, a, false, b, (short)0, c, false, false);
    }

    if (act) {
        int r0 = rowBase + (hi ? 8 : 0);
        float* outp = gpart + split * GPS;
        #pragma unroll
        for (int j = 0; j < 8; ++j) outp[r0 + j] = c[j];
    }
}

// ---------------------------------------------------------------------------
// Encoder elementwise GRU update (8 blocks x 256 = 2048 lanes).
// gi = u*A + B (collapsed); gh = sum of 4 K-split partials.
// carry h = sigmoid(hp); export bf16(h) and bf16(hp) (y = last raw hp).
// ---------------------------------------------------------------------------
__global__ __launch_bounds__(256) void gru_update_enc(
    const float* __restrict__ gp,     // 4 partial buffers, stride GPS
    const float* __restrict__ A, const float* __restrict__ B,
    const float* __restrict__ bhh,
    const float* __restrict__ hist, int t,
    float* __restrict__ h, unsigned short* __restrict__ hb,
    unsigned short* __restrict__ xb)
{
    float u = hist[t];
    int i = blockIdx.x * 256 + threadIdx.x;          // 0..2047
    float gir = fmaf(u, A[i],          B[i]);
    float giz = fmaf(u, A[Hh + i],     B[Hh + i]);
    float gic = fmaf(u, A[2 * Hh + i], B[2 * Hh + i]);
    float ghr = gsum(gp, i)          + bhh[i];
    float ghz = gsum(gp, Hh + i)     + bhh[Hh + i];
    float ghc = gsum(gp, 2 * Hh + i) + bhh[2 * Hh + i];
    float r  = sigm(gir + ghr);
    float z  = sigm(giz + ghz);
    float cc = tanhf(gic + r * ghc);
    float hp = (1.f - z) * cc + z * h[i];
    float hs = sigm(hp);
    h[i]  = hs;
    hb[i] = f2bf(hs);
    xb[i] = f2bf(hp);     // only the final step's value is consumed
}

// ---------------------------------------------------------------------------
// Decoder elementwise GRU update + beta = sigmoid(hp . Wa + ba) reduction.
// Rows [0,6144) of each partial = gi (Wih@x), rows [6144,12288) = gh (Whh@h).
// ---------------------------------------------------------------------------
__global__ __launch_bounds__(256) void gru_update_dec(
    const float* __restrict__ gp,
    const float* __restrict__ bih, const float* __restrict__ bhh,
    const float* __restrict__ Wa,  const float* __restrict__ ba,
    float* __restrict__ h, unsigned short* __restrict__ hb,
    unsigned short* __restrict__ xb,
    float* __restrict__ betas, int t)
{
    __shared__ float red[256];
    float acc = 0.f;
    for (int i = threadIdx.x; i < Hh; i += 256) {
        float gir = gsum(gp, i)              + bih[i];
        float giz = gsum(gp, Hh + i)         + bih[Hh + i];
        float gic = gsum(gp, 2 * Hh + i)     + bih[2 * Hh + i];
        float ghr = gsum(gp, R3H + i)          + bhh[i];
        float ghz = gsum(gp, R3H + Hh + i)     + bhh[Hh + i];
        float ghc = gsum(gp, R3H + 2 * Hh + i) + bhh[2 * Hh + i];
        float r  = sigm(gir + ghr);
        float z  = sigm(giz + ghz);
        float cc = tanhf(gic + r * ghc);
        float hp = (1.f - z) * cc + z * h[i];
        h[i]  = hp;
        hb[i] = f2bf(hp);
        xb[i] = f2bf(hp);
        acc = fmaf(hp, Wa[i], acc);
    }
    red[threadIdx.x] = acc;
    __syncthreads();
    for (int s = 128; s; s >>= 1) {
        if (threadIdx.x < (unsigned)s) red[threadIdx.x] += red[threadIdx.x + s];
        __syncthreads();
    }
    if (threadIdx.x == 0) betas[t] = sigm(red[0] + ba[0]);
}

// ---------------------------------------------------------------------------
// Scalar SIRD rollout. out = [x_final(5) ; x_foo(29*5)]
// ---------------------------------------------------------------------------
__global__ void sird_final(const float* __restrict__ sird,
                           const float* __restrict__ betas,
                           const int* __restrict__ tstep,
                           float* __restrict__ out)
{
    if (threadIdx.x != 0 || blockIdx.x != 0) return;
    float s = sird[0], i = sird[1], r = sird[2], d = sird[3], n = sird[4];
    #pragma unroll 1
    for (int k = 0; k < FSTEPS; ++k) {
        float beta = betas[k];
        float si = s * i * beta / n;
        float ir = i * 0.05f, id = i * 0.01f;
        float s2 = s - si, i2 = i + si - ir - id;
        r += ir; d += id; s = s2; i = i2;
    }
    float bl = betas[FSTEPS - 1];
    int T = tstep[0] - 1;
    #pragma unroll 1
    for (int k = 0; k < T; ++k) {
        float si = s * i * bl / n;
        float ir = i * 0.05f, id = i * 0.01f;
        float s2 = s - si, i2 = i + si - ir - id;
        r += ir; d += id; s = s2; i = i2;
        int o = 5 + k * 5;
        out[o + 0] = s; out[o + 1] = i; out[o + 2] = r; out[o + 3] = d; out[o + 4] = n;
    }
    out[0] = s; out[1] = i; out[2] = r; out[3] = d; out[4] = n;
}

// ---------------------------------------------------------------------------
extern "C" void kernel_launch(void* const* d_in, const int* in_sizes, int n_in,
                              void* d_out, int out_size, void* d_ws, size_t ws_size,
                              hipStream_t stream)
{
    (void)in_sizes; (void)n_in; (void)out_size; (void)ws_size;

    const float* hist  = (const float*)d_in[0];
    const float* sird  = (const float*)d_in[1];
    const int*   tstep = (const int*)d_in[2];
    const float* Wb    = (const float*)d_in[3];
    const float* bb    = (const float*)d_in[4];
    const float* eWih  = (const float*)d_in[5];
    const float* eWhh  = (const float*)d_in[6];
    const float* ebih  = (const float*)d_in[7];
    const float* ebhh  = (const float*)d_in[8];
    const float* dWih  = (const float*)d_in[9];
    const float* dWhh  = (const float*)d_in[10];
    const float* dbih  = (const float*)d_in[11];
    const float* dbhh  = (const float*)d_in[12];
    const float* Wa    = (const float*)d_in[13];
    const float* ba    = (const float*)d_in[14];

    const size_t WN = (size_t)R3H * Hh;              // 12,582,912 weights
    unsigned short* eWhh_b = (unsigned short*)d_ws;
    unsigned short* dWih_b = eWhh_b + WN;
    unsigned short* dWhh_b = dWih_b + WN;
    float* Aenc  = (float*)(dWhh_b + WN);
    float* Benc  = Aenc + R3H;
    float* gpart = Benc + R3H;                       // [KSPLIT * GPS]
    float* hst   = gpart + KSPLIT * GPS;             // [2048] fp32 carry
    unsigned short* hbf = (unsigned short*)(hst + Hh);  // [2048] bf16 h
    unsigned short* xbf = hbf + Hh;                     // [2048] bf16 x
    float* betas = (float*)(xbf + Hh);               // [90]

    int cb = (int)((WN + 255) / 256);
    f32_to_bf16_k<<<cb, 256, 0, stream>>>(eWhh, eWhh_b, (int)WN);
    f32_to_bf16_k<<<cb, 256, 0, stream>>>(dWih, dWih_b, (int)WN);
    f32_to_bf16_k<<<cb, 256, 0, stream>>>(dWhh, dWhh_b, (int)WN);
    enc_input_proj<<<R3H / 8, 256, 0, stream>>>(eWih, Wb, bb, ebih, Aenc, Benc);
    init_state<<<8, 256, 0, stream>>>(hst, hbf, xbf);

    // Encoder: one mat-vec (Whh@h) per step — Wih projection is collapsed.
    // 384 strips * 4 K-splits = 1536 waves -> 192 blocks.
    for (int t = 0; t < TSTEPS; ++t) {
        matvec_bf16_wmma<<<192, 256, 0, stream>>>(eWhh_b, hbf, eWhh_b, hbf, gpart);
        gru_update_enc<<<8, 256, 0, stream>>>(gpart, Aenc, Benc, ebhh, hist, t,
                                              hst, hbf, xbf);
    }
    // Decoder: stacked [Wih@x ; Whh@h] -> 768 strips * 4 splits -> 384 blocks.
    for (int t = 0; t < FSTEPS; ++t) {
        matvec_bf16_wmma<<<384, 256, 0, stream>>>(dWih_b, xbf, dWhh_b, hbf, gpart);
        gru_update_dec<<<1, 256, 0, stream>>>(gpart, dbih, dbhh, Wa, ba,
                                              hst, hbf, xbf, betas, t);
    }
    sird_final<<<1, 1, 0, stream>>>(sird, betas, tstep, (float*)d_out);
}
